// myGPT2Block_44667659879006
// MI455X (gfx1250) — matrix operand order repaired and checked by reading, not compiled, not run on hardware
//
#include <hip/hip_runtime.h>

// ---------------------------------------------------------------------------
// GPT-2 block for MI455X (gfx1250, wave32, WMMA). All GEMMs + attention run on
// v_wmma_f32_16x16x32_bf16 with fp32 accumulation; LN/bias/scale/mask/softmax/
// LeakyReLU/residual fused into epilogues. GEMM is software-pipelined
// (regs <-> LDS double use) with global_prefetch_b8 distance-2 prefetch.
// ---------------------------------------------------------------------------

typedef __attribute__((ext_vector_type(16))) __bf16 v16bf;
typedef __attribute__((ext_vector_type(8)))  float  v8f;

#define DEV __device__ __forceinline__

constexpr int BS    = 2;
constexpr int SEQ   = 2048;
constexpr int DM    = 1024;
constexpr int NH    = 16;
constexpr int HD    = 64;
constexpr int INNER = 4096;
constexpr int MROWS = BS * SEQ;          // 4096
constexpr int LDSS  = 72;                 // padded LDS row stride (elems): 144B, 16B-aligned

DEV unsigned short f2bf(float x) {
  unsigned int u = __float_as_uint(x);
  u += 0x7FFFu + ((u >> 16) & 1u);
  return (unsigned short)(u >> 16);
}

DEV v8f vzero() {
  v8f z;
#pragma unroll
  for (int i = 0; i < 8; ++i) z[i] = 0.f;
  return z;
}

union FragU { v16bf v; unsigned int u[8]; };

// A fragment (16x32 bf16): lane holds row M = lane%16.
// K(d) = (lane&16 ? 16 : 0) + d*2  -> contiguous dword pairs.
DEV v16bf ld_a_frag(const unsigned short* rowptr, int lane) {
  const unsigned int* p = (const unsigned int*)rowptr;
  const int kh = (lane & 16) ? 8 : 0;    // dwords
  FragU f;
#pragma unroll
  for (int d = 0; d < 8; ++d) f.u[d] = p[kh + d];
  return f.v;
}

// B fragment (32x16 bf16): lane holds col N = lane%16; lanes 0-15: K=0..15,
// lanes 16-31: K=16..31, VGPR d -> K = base + 2d..2d+1 -> contiguous dwords.
DEV v16bf ld_b_frag(const unsigned short* rowptr, int lane) {
  const unsigned int* p = (const unsigned int*)rowptr;
  const int base = (lane & 16) ? 8 : 0;  // dwords
  FragU f;
#pragma unroll
  for (int d = 0; d < 8; ++d) f.u[d] = p[base + d];
  return f.v;
}

DEV v8f wmma_bf16(v16bf a, v16bf b, v8f c) {
  return __builtin_amdgcn_wmma_f32_16x16x32_bf16(false, a, false, b, (short)0, c,
                                                 false, false);
}

// ---------------------------------------------------------------------------
// fp32 -> bf16 conversion (weights)
// ---------------------------------------------------------------------------
__global__ __launch_bounds__(256) void cvt_f32_bf16(const float* __restrict__ in,
                                                    unsigned short* __restrict__ out,
                                                    int n) {
  int i = blockIdx.x * 256 + threadIdx.x;
  if (i < n) out[i] = f2bf(in[i]);
}

// ---------------------------------------------------------------------------
// Fused LayerNorm -> bf16 (one row of D=1024 per 256-thread block)
// ---------------------------------------------------------------------------
__global__ __launch_bounds__(256) void layernorm_to_bf16(const float* __restrict__ x,
                                                         const float* __restrict__ g,
                                                         const float* __restrict__ b,
                                                         unsigned short* __restrict__ out) {
  __shared__ float red[16];
  const int row = blockIdx.x, tid = threadIdx.x;
  const float* xr = x + (size_t)row * DM;
  float v[4], s = 0.f, ss = 0.f;
#pragma unroll
  for (int i = 0; i < 4; ++i) { v[i] = xr[tid + i * 256]; s += v[i]; ss += v[i] * v[i]; }
#pragma unroll
  for (int m = 16; m >= 1; m >>= 1) { s += __shfl_xor(s, m, 32); ss += __shfl_xor(ss, m, 32); }
  if ((tid & 31) == 0) { red[tid >> 5] = s; red[8 + (tid >> 5)] = ss; }
  __syncthreads();
  float S = 0.f, SS = 0.f;
#pragma unroll
  for (int w = 0; w < 8; ++w) { S += red[w]; SS += red[8 + w]; }
  const float mu  = S * (1.f / DM);
  const float inv = rsqrtf(SS * (1.f / DM) - mu * mu + 1e-5f);
#pragma unroll
  for (int i = 0; i < 4; ++i) {
    int c = tid + i * 256;
    out[(size_t)row * DM + c] = f2bf((v[i] - mu) * inv * g[c] + b[c]);
  }
}

// ---------------------------------------------------------------------------
// Pipelined tiled WMMA GEMM: C[128x128] per block, 8 waves (4x2), wave = 32x64.
// K-step 64 (16 wmma per stage). A row-major [M,K] bf16; B row-major [K,N]
// bf16 (ldb), transposed into LDS so fragments are contiguous ds_load_b32.
// Next K tile is loaded to registers while wmma consumes LDS; distance-2 tile
// prefetched with global_prefetch_b8. Epilogues:
//  0 = Q: (+bias)*0.125 -> bf16 head layout [B,H,S,HD]
//  1 = K: (+bias)       -> bf16 head layout [B,H,S,HD]
//  2 = V:               -> bf16 transposed head layout [B,H,HD,S]
//  3 = +residual        -> fp32 row-major [M,DM]
//  4 = LeakyReLU(0.01)  -> bf16 row-major [M,INNER]
// ---------------------------------------------------------------------------
template <int EPI>
__global__ __launch_bounds__(256)
void gemm_bf16_k(const unsigned short* __restrict__ A, int lda,
                 const unsigned short* __restrict__ Bm, int ldb, int K,
                 const float* __restrict__ bias, const float* __restrict__ resid,
                 float* __restrict__ outf, unsigned short* __restrict__ outh) {
  __shared__ unsigned short sA[128 * LDSS];   // [m][k]  128 x 64 (+pad)
  __shared__ unsigned short sB[128 * LDSS];   // [n][k]  (transposed) 128 x 64 (+pad)
  const int tid  = threadIdx.x;
  const int lane = tid & 31, wave = tid >> 5;
  const int wm = wave & 3, wn = wave >> 2;
  const int m0 = blockIdx.y * 128, n0 = blockIdx.x * 128;
  const int l15 = lane & 15;

  v8f acc[2][4];
#pragma unroll
  for (int i = 0; i < 2; ++i)
#pragma unroll
    for (int j = 0; j < 4; ++j) acc[i][j] = vzero();

  uint4        ra[4];      // A staging: 128*64 elems / 256 thr = 32 elems = 4 x uint4
  unsigned int rb[8][2];   // B staging: 8 dword-pair groups per thread

  auto gload = [&](int kt) {
#pragma unroll
    for (int i = 0; i < 4; ++i) {
      int u = tid * 4 + i;                 // uint4 id; 8 per row
      int row = u >> 3, kq = u & 7;
      ra[i] = *(const uint4*)(A + (size_t)(m0 + row) * lda + kt + kq * 8);
    }
#pragma unroll
    for (int i = 0; i < 8; ++i) {
      int g = tid + i * 256;               // pair-group id (2048 total)
      int n = (g & 63) * 2, k = (g >> 6) * 2;
      rb[i][0] = *(const unsigned int*)(Bm + (size_t)(kt + k) * ldb + n0 + n);
      rb[i][1] = *(const unsigned int*)(Bm + (size_t)(kt + k + 1) * ldb + n0 + n);
    }
  };
  auto lstore = [&]() {
#pragma unroll
    for (int i = 0; i < 4; ++i) {
      int u = tid * 4 + i;
      int row = u >> 3, kq = u & 7;
      *(uint4*)(&sA[row * LDSS + kq * 8]) = ra[i];
    }
#pragma unroll
    for (int i = 0; i < 8; ++i) {
      int g = tid + i * 256;
      int n = (g & 63) * 2, k = (g >> 6) * 2;
      unsigned int w0 = rb[i][0], w1 = rb[i][1];
      *(unsigned int*)(&sB[(size_t)n * LDSS + k])       = (w0 & 0xFFFFu) | (w1 << 16);
      *(unsigned int*)(&sB[(size_t)(n + 1) * LDSS + k]) = (w0 >> 16) | (w1 & 0xFFFF0000u);
    }
  };

  gload(0);
  for (int kt = 0; kt < K; kt += 64) {
    __syncthreads();                       // previous compute done reading LDS
    lstore();
    __syncthreads();                       // tile visible to all waves
    if (kt + 64 < K) {
      gload(kt + 64);                      // overlap next tile load with wmma
      if (kt + 128 < K) {                  // distance-2 L2/WGP$ prefetch
        __builtin_prefetch(A + (size_t)(m0 + (tid & 127)) * lda + kt + 128, 0, 1);
        __builtin_prefetch(Bm + (size_t)(kt + 128 + (tid & 63)) * ldb + n0 + (tid >> 6) * 32, 0, 1);
      }
    }
#pragma unroll
    for (int kc = 0; kc < 64; kc += 32) {
      v16bf af0 = ld_a_frag(&sA[(size_t)(wm * 32 + l15) * LDSS + kc], lane);
      v16bf af1 = ld_a_frag(&sA[(size_t)(wm * 32 + 16 + l15) * LDSS + kc], lane);
#pragma unroll
      for (int nf = 0; nf < 4; ++nf) {
        v16bf bf = ld_b_frag(&sB[(size_t)(wn * 64 + nf * 16 + l15) * LDSS + kc], lane);
        acc[0][nf] = wmma_bf16(af0, bf, acc[0][nf]);
        acc[1][nf] = wmma_bf16(af1, bf, acc[1][nf]);
      }
    }
  }

  const int rbase = m0 + wm * 32 + ((lane & 16) ? 8 : 0);
  const int cbase = n0 + wn * 64 + l15;
#pragma unroll
  for (int mf = 0; mf < 2; ++mf)
#pragma unroll
    for (int nf = 0; nf < 4; ++nf)
#pragma unroll
      for (int r = 0; r < 8; ++r) {
        int row = rbase + mf * 16 + r;
        int col = cbase + nf * 16;
        float v = acc[mf][nf][r];
        if constexpr (EPI == 0 || EPI == 1) {
          v += bias[col];
          if constexpr (EPI == 0) v *= 0.125f;   // 1/sqrt(HD)
          int b = row >> 11, s = row & (SEQ - 1), hh = col >> 6, hd = col & 63;
          outh[(((size_t)(b * NH + hh) * SEQ + s) << 6) + hd] = f2bf(v);
        } else if constexpr (EPI == 2) {
          int b = row >> 11, s = row & (SEQ - 1), hh = col >> 6, hd = col & 63;
          outh[(((size_t)(b * NH + hh) * HD + hd) << 11) + s] = f2bf(v);
        } else if constexpr (EPI == 3) {
          size_t idx = (size_t)row * DM + col;
          outf[idx] = resid[idx] + v;
        } else {
          outh[(size_t)row * INNER + col] = f2bf(v >= 0.f ? v : v * 0.01f);
        }
      }
}

// ---------------------------------------------------------------------------
// Flash attention: one wave per 16-query tile, 32-key tiles, causal online
// softmax in registers (row stats reduce in 16-lane halves), P staged through
// per-wave LDS to convert C-layout -> A-layout. Next key tile prefetched.
// q: [B,H,S,HD] (pre-scaled), k: [B,H,S,HD], vt: [B,H,HD,S], out: [B,S,D] bf16
// ---------------------------------------------------------------------------
__global__ __launch_bounds__(128)
void flash_attn(const unsigned short* __restrict__ qbuf,
                const unsigned short* __restrict__ kbuf,
                const unsigned short* __restrict__ vtbuf,
                unsigned short* __restrict__ attn) {
  __shared__ unsigned short pstage[4][16 * 32];
  const int lane = threadIdx.x & 31, wave = threadIdx.x >> 5;
  const int bh   = blockIdx.x >> 5;        // head index (B*H = 32)
  const int qblk = blockIdx.x & 31;        // 64-query block within sequence
  const int q0   = qblk * 64 + wave * 16;
  const unsigned short* qh = qbuf  + (size_t)bh * SEQ * HD;
  const unsigned short* kh = kbuf  + (size_t)bh * SEQ * HD;
  const unsigned short* vh = vtbuf + (size_t)bh * HD * SEQ;
  const int l15  = lane & 15;
  const int rsub = (lane & 16) ? 8 : 0;

  v16bf qa0 = ld_a_frag(qh + (size_t)(q0 + l15) * HD, lane);
  v16bf qa1 = ld_a_frag(qh + (size_t)(q0 + l15) * HD + 32, lane);

  v8f o0 = vzero(), o1 = vzero(), o2 = vzero(), o3 = vzero();
  float mrun[8], lrun[8];
#pragma unroll
  for (int r = 0; r < 8; ++r) { mrun[r] = -1e30f; lrun[r] = 0.f; }

  unsigned short* ps = pstage[wave];
  const int kend = q0 + 16;
  for (int kb = 0; kb < kend; kb += 32) {
    if (kb + 32 < kend) {                  // prefetch next key/value tile
      __builtin_prefetch(kh + (size_t)(kb + 32 + lane) * HD, 0, 1);
      __builtin_prefetch(vh + (size_t)lane * SEQ + kb + 32, 0, 1);
      __builtin_prefetch(vh + (size_t)(lane + 32) * SEQ + kb + 32, 0, 1);
    }
    const unsigned short* kr0 = kh + (size_t)(kb + l15) * HD;
    const unsigned short* kr1 = kr0 + 16 * HD;
    v8f s0 = vzero(), s1 = vzero();
    s0 = wmma_bf16(qa0, ld_b_frag(kr0, lane), s0);
    s0 = wmma_bf16(qa1, ld_b_frag(kr0 + 32, lane), s0);
    s1 = wmma_bf16(qa0, ld_b_frag(kr1, lane), s1);
    s1 = wmma_bf16(qa1, ld_b_frag(kr1 + 32, lane), s1);

    if (kb + 31 > q0) {                    // causal mask (arithmetic, no diverge)
#pragma unroll
      for (int r = 0; r < 8; ++r) {
        int rowq = q0 + rsub + r;
        if (kb + l15 > rowq)      s0[r] = -1e30f;
        if (kb + 16 + l15 > rowq) s1[r] = -1e30f;
      }
    }
#pragma unroll
    for (int r = 0; r < 8; ++r) {
      float tmax = fmaxf(s0[r], s1[r]);
#pragma unroll
      for (int m = 8; m >= 1; m >>= 1) tmax = fmaxf(tmax, __shfl_xor(tmax, m, 32));
      float mnew = fmaxf(mrun[r], tmax);
      float corr = __expf(mrun[r] - mnew);
      float p0 = __expf(s0[r] - mnew);
      float p1 = __expf(s1[r] - mnew);
      float psum = p0 + p1;
#pragma unroll
      for (int m = 8; m >= 1; m >>= 1) psum += __shfl_xor(psum, m, 32);
      lrun[r] = lrun[r] * corr + psum;
      mrun[r] = mnew;
      o0[r] *= corr; o1[r] *= corr; o2[r] *= corr; o3[r] *= corr;
      ps[(rsub + r) * 32 + l15]      = f2bf(p0);
      ps[(rsub + r) * 32 + 16 + l15] = f2bf(p1);
    }
    // per-wave LDS; DS ops execute in order within a wave — just stop the
    // compiler from reordering the typed accesses.
    asm volatile("" ::: "memory");
    v16bf pf = ld_a_frag(ps + l15 * 32, lane);
    asm volatile("" ::: "memory");
    const unsigned short* vr = vh + (size_t)l15 * SEQ + kb;
    o0 = wmma_bf16(pf, ld_b_frag(vr, lane), o0);
    o1 = wmma_bf16(pf, ld_b_frag(vr + 16 * SEQ, lane), o1);
    o2 = wmma_bf16(pf, ld_b_frag(vr + 32 * SEQ, lane), o2);
    o3 = wmma_bf16(pf, ld_b_frag(vr + 48 * SEQ, lane), o3);
  }

  const int b = bh >> 4, hh = bh & 15;
#pragma unroll
  for (int r = 0; r < 8; ++r) {
    float invl = 1.f / lrun[r];
    size_t ro = ((size_t)(b * SEQ + q0 + rsub + r)) * DM + hh * HD + l15;
    attn[ro + 0]  = f2bf(o0[r] * invl);
    attn[ro + 16] = f2bf(o1[r] * invl);
    attn[ro + 32] = f2bf(o2[r] * invl);
    attn[ro + 48] = f2bf(o3[r] * invl);
  }
}

// ---------------------------------------------------------------------------
// Host orchestration
// ---------------------------------------------------------------------------
extern "C" void kernel_launch(void* const* d_in, const int* in_sizes, int n_in,
                              void* d_out, int out_size, void* d_ws, size_t ws_size,
                              hipStream_t stream) {
  const float* x    = (const float*)d_in[0];
  const float* ln1g = (const float*)d_in[1];
  const float* ln1b = (const float*)d_in[2];
  const float* ln2g = (const float*)d_in[3];
  const float* ln2b = (const float*)d_in[4];
  const float* Wqk  = (const float*)d_in[5];
  const float* bqk  = (const float*)d_in[6];
  const float* Wv   = (const float*)d_in[7];
  const float* Wo   = (const float*)d_in[8];
  const float* Wfc  = (const float*)d_in[9];
  const float* Wpj  = (const float*)d_in[10];
  float* out = (float*)d_out;

  char* ws = (char*)d_ws;
  const size_t MB = 1024 * 1024;
  unsigned short* wqk_h = (unsigned short*)(ws + 0);        // 4 MB
  unsigned short* wv_h  = (unsigned short*)(ws + 4  * MB);  // 2 MB
  unsigned short* wo_h  = (unsigned short*)(ws + 6  * MB);  // 2 MB
  unsigned short* wfc_h = (unsigned short*)(ws + 8  * MB);  // 8 MB
  unsigned short* wpj_h = (unsigned short*)(ws + 16 * MB);  // 8 MB
  unsigned short* h_h   = (unsigned short*)(ws + 24 * MB);  // 8 MB
  unsigned short* q_h   = (unsigned short*)(ws + 32 * MB);  // 8 MB
  unsigned short* k_h   = (unsigned short*)(ws + 40 * MB);  // 8 MB
  unsigned short* vt_h  = (unsigned short*)(ws + 48 * MB);  // 8 MB
  unsigned short* at_h  = (unsigned short*)(ws + 56 * MB);  // 8 MB
  float*          x1    = (float*)        (ws + 64 * MB);   // 16 MB
  unsigned short* h2_h  = (unsigned short*)(ws + 80 * MB);  // 8 MB
  unsigned short* a_h   = (unsigned short*)(ws + 88 * MB);  // 32 MB -> 120 MB total

  auto cvt = [&](const float* src, unsigned short* dst, int n) {
    cvt_f32_bf16<<<(n + 255) / 256, 256, 0, stream>>>(src, dst, n);
  };
  cvt(Wqk, wqk_h, DM * 2 * DM);
  cvt(Wv,  wv_h,  DM * DM);
  cvt(Wo,  wo_h,  DM * DM);
  cvt(Wfc, wfc_h, DM * INNER);
  cvt(Wpj, wpj_h, INNER * DM);

  layernorm_to_bf16<<<MROWS, 256, 0, stream>>>(x, ln1g, ln1b, h_h);

  dim3 g8(DM / 128, MROWS / 128);       // (8, 32)
  dim3 g32(INNER / 128, MROWS / 128);   // (32, 32)
  gemm_bf16_k<0><<<g8, 256, 0, stream>>>(h_h, DM, wqk_h,      2 * DM, DM, bqk,      nullptr, nullptr, q_h);
  gemm_bf16_k<1><<<g8, 256, 0, stream>>>(h_h, DM, wqk_h + DM, 2 * DM, DM, bqk + DM, nullptr, nullptr, k_h);
  gemm_bf16_k<2><<<g8, 256, 0, stream>>>(h_h, DM, wv_h,       DM,     DM, nullptr,  nullptr, nullptr, vt_h);

  flash_attn<<<BS * NH * (SEQ / 64), 128, 0, stream>>>(q_h, k_h, vt_h, at_h);

  gemm_bf16_k<3><<<g8, 256, 0, stream>>>(at_h, DM, wo_h, DM, DM, nullptr, x, x1, nullptr);
  layernorm_to_bf16<<<MROWS, 256, 0, stream>>>(x1, ln2g, ln2b, h2_h);
  gemm_bf16_k<4><<<g32, 256, 0, stream>>>(h2_h, DM, wfc_h, INNER, DM, nullptr, nullptr, nullptr, a_h);
  gemm_bf16_k<3><<<g8, 256, 0, stream>>>(a_h, INNER, wpj_h, DM, INNER, nullptr, x1, out, nullptr);
}
